// KHop_66546223284512
// MI455X (gfx1250) — matrix operand order repaired and checked by reading, not compiled, not run on hardware
//
#include <hip/hip_runtime.h>
#include <hip/hip_bf16.h>
#include <math.h>

typedef __attribute__((ext_vector_type(16))) __bf16 v16bf;
typedef __attribute__((ext_vector_type(8)))  float  v8f;

// ---------------------------------------------------------------------------
// WMMA helper: D = A(16x32 bf16) * B(32x16 bf16) + C(16x16 f32)
// ---------------------------------------------------------------------------
__device__ __forceinline__ v8f wmma_bf16(v16bf a, v16bf b, v8f c) {
  return __builtin_amdgcn_wmma_f32_16x16x32_bf16(
      /*neg_a=*/false, a, /*neg_b=*/false, b,
      /*c_mod=*/(short)0, c, /*reuse_a=*/false, /*reuse_b=*/false);
}

// A-matrix K-permutation within each 32-block so each lane's 16 bf16 are
// contiguous in LDS.  Lanes 0-15 need K {0..7,16..23}, lanes 16-31 need
// K {8..15,24..31}; store them as [0..7,16..23 | 8..15,24..31].
__device__ __forceinline__ int permk(int k) {
  int j = k & 31, b = k & ~31;
  int jp = (j < 8) ? j : (j < 16) ? (j + 8) : (j < 24) ? (j - 8) : j;
  return b + jp;
}

// ---------------------------------------------------------------------------
// Weight packing: fp32 [K x Nout] row-major -> bf16 B-fragments.
// Fragment (kt,nt): lane l, elem i -> (K = kt*32 + 8*(l>=16) + (i<8 ? i : i+8), N = nt*16 + (l&15))
// ---------------------------------------------------------------------------
__global__ void pack_w_kernel(const float* __restrict__ W, __bf16* __restrict__ out,
                              int K, int Nout, int Ktiles, int Ntiles) {
  int idx = blockIdx.x * blockDim.x + threadIdx.x;
  int total = Ktiles * Ntiles * 512;
  if (idx >= total) return;
  int i    = idx & 15;
  int lane = (idx >> 4) & 31;
  int fr   = idx >> 9;
  int nt   = fr % Ntiles, kt = fr / Ntiles;
  int k = kt * 32 + ((i < 8) ? i : (i + 8)) + ((lane >= 16) ? 8 : 0);
  int n = nt * 16 + (lane & 15);
  float v = (k < K) ? W[k * Nout + n] : 0.0f;
  out[idx] = (__bf16)v;
}

__global__ void zero_kernel(float* __restrict__ p, int n) {
  int i = blockIdx.x * blockDim.x + threadIdx.x;
  if (i < n) p[i] = 0.0f;
}

// ---------------------------------------------------------------------------
// Edge features: ef[E,8] = [diff3,diff4,diff5,dist,ux,uy,uz,0], mask[E]
// ---------------------------------------------------------------------------
__global__ void efeat_kernel(const float* __restrict__ x, const int* __restrict__ send,
                             const int* __restrict__ recv, float* __restrict__ ef,
                             float* __restrict__ mask, int E) {
  int e = blockIdx.x * blockDim.x + threadIdx.x;
  if (e >= E) return;
  int s = send[e], r = recv[e];
  float d[6];
#pragma unroll
  for (int i = 0; i < 6; ++i) d[i] = x[r * 6 + i] - x[s * 6 + i];
  float dist = sqrtf(d[0] * d[0] + d[1] * d[1] + d[2] * d[2]);
  float inv = (dist > 0.0f) ? (1.0f / dist) : 0.0f;
  ef[e * 8 + 0] = d[3];
  ef[e * 8 + 1] = d[4];
  ef[e * 8 + 2] = d[5];
  ef[e * 8 + 3] = dist;
  ef[e * 8 + 4] = d[0] * inv;
  ef[e * 8 + 5] = d[1] * inv;
  ef[e * 8 + 6] = d[2] * inv;
  ef[e * 8 + 7] = 0.0f;
  mask[e] = (x[s * 6 + 3] <= x[r * 6 + 3]) ? 1.0f : 0.0f;
}

// ---------------------------------------------------------------------------
// Message hop: per (MTILES*16)-edge tile, MLP([h_recv,h_send,e]) -> masked
// scatter-add.  Weight fragments are reused across MTILES A sub-tiles to
// raise WMMA arithmetic intensity vs L2 weight traffic.
// Block = 256 threads = 8 waves.
// ---------------------------------------------------------------------------
template <int D, int KPAD, int MTILES>
__global__ void msg_hop_kernel(const float* __restrict__ h, const float* __restrict__ ef,
                               const float* __restrict__ mask, const int* __restrict__ send,
                               const int* __restrict__ recv,
                               const v16bf* __restrict__ w1f, const float* __restrict__ b1,
                               const v16bf* __restrict__ w2f, const float* __restrict__ b2,
                               float* __restrict__ out, int E) {
  constexpr int NH = 256, NOUT = 128;
  constexpr int KT1 = KPAD / 32, NT1 = NH / 16;
  constexpr int KT2 = NH / 32,  NT2 = NOUT / 16;
  constexpr int M = MTILES * 16;
  __shared__ __attribute__((aligned(32))) __bf16 Abuf[M * KPAD];
  __shared__ __attribute__((aligned(32))) __bf16 Hbuf[M * NH];

  int e0 = blockIdx.x * M;
  int wave = threadIdx.x >> 5, lane = threadIdx.x & 31;

  // Stage 1: gather [h[recv], h[send], e] into LDS (bf16, A-permuted).
  // One wave per row; lanes sweep columns (coalesced row reads, no div/mod).
  for (int row = wave; row < M; row += 8) {
    int e = e0 + row;
    bool ok = (e < E);
    int rv = ok ? recv[e] : 0;
    int sv = ok ? send[e] : 0;
    for (int k = lane; k < KPAD; k += 32) {
      float v = 0.0f;
      if (ok) {
        if (k < D)              v = h[rv * D + k];
        else if (k < 2 * D)     v = h[sv * D + (k - D)];
        else if (k < 2 * D + 7) v = ef[e * 8 + (k - 2 * D)];
      }
      Abuf[row * KPAD + permk(k)] = (__bf16)v;
    }
  }
  __syncthreads();

  int kh = lane >> 4, mrow = lane & 15;

  // Stage 2: GEMM1 (KPAD -> 256); each wave owns 2 N-tiles, MTILES M-subtiles.
  v8f acc[MTILES][2];
#pragma unroll
  for (int mt = 0; mt < MTILES; ++mt) {
    acc[mt][0] = (v8f){0, 0, 0, 0, 0, 0, 0, 0};
    acc[mt][1] = (v8f){0, 0, 0, 0, 0, 0, 0, 0};
  }
#pragma unroll
  for (int kt = 0; kt < KT1; ++kt) {
    v16bf b0 = w1f[(kt * NT1 + wave * 2 + 0) * 32 + lane];
    v16bf b1v = w1f[(kt * NT1 + wave * 2 + 1) * 32 + lane];
#pragma unroll
    for (int mt = 0; mt < MTILES; ++mt) {
      v16bf a = *(const v16bf*)&Abuf[(mt * 16 + mrow) * KPAD + kt * 32 + kh * 16];
      acc[mt][0] = wmma_bf16(a, b0, acc[mt][0]);
      acc[mt][1] = wmma_bf16(a, b1v, acc[mt][1]);
    }
  }
  float bias10 = b1[(wave * 2 + 0) * 16 + mrow];
  float bias11 = b1[(wave * 2 + 1) * 16 + mrow];
#pragma unroll
  for (int mt = 0; mt < MTILES; ++mt) {
#pragma unroll
    for (int t = 0; t < 2; ++t) {
      int n = (wave * 2 + t) * 16 + mrow;
      float bb = t ? bias11 : bias10;
#pragma unroll
      for (int r = 0; r < 8; ++r) {
        int m = mt * 16 + r + kh * 8;
        float v = acc[mt][t][r] + bb;
        v = v > 0.0f ? v : 0.0f;
        Hbuf[m * NH + permk(n)] = (__bf16)v;
      }
    }
  }
  __syncthreads();

  // Stage 3: GEMM2 (256 -> 128); each wave owns 1 N-tile; relu*mask; scatter.
  v8f acc2[MTILES];
#pragma unroll
  for (int mt = 0; mt < MTILES; ++mt) acc2[mt] = (v8f){0, 0, 0, 0, 0, 0, 0, 0};
#pragma unroll
  for (int kt = 0; kt < KT2; ++kt) {
    v16bf b = w2f[(kt * NT2 + wave) * 32 + lane];
#pragma unroll
    for (int mt = 0; mt < MTILES; ++mt) {
      v16bf a = *(const v16bf*)&Hbuf[(mt * 16 + mrow) * NH + kt * 32 + kh * 16];
      acc2[mt] = wmma_bf16(a, b, acc2[mt]);
    }
  }
  float bias2 = b2[wave * 16 + mrow];
  int ncol = wave * 16 + mrow;
#pragma unroll
  for (int mt = 0; mt < MTILES; ++mt) {
#pragma unroll
    for (int r = 0; r < 8; ++r) {
      int m = mt * 16 + r + kh * 8;
      int e = e0 + m;
      if (e < E) {
        float v = acc2[mt][r] + bias2;
        v = v > 0.0f ? v : 0.0f;
        v *= mask[e];
        atomicAdd(&out[recv[e] * NOUT + ncol], v);
      }
    }
  }
}

// ---------------------------------------------------------------------------
// Node update MLP: h[N,128] -> relu(128->128) -> relu(128->64), 64 nodes/block
// ---------------------------------------------------------------------------
__global__ void upd_kernel(const float* __restrict__ h,
                           const v16bf* __restrict__ w1f, const float* __restrict__ b1,
                           const v16bf* __restrict__ w2f, const float* __restrict__ b2,
                           float* __restrict__ out, int nNodes) {
  constexpr int M = 64;
  __shared__ __attribute__((aligned(32))) __bf16 Abuf[M * 128];
  __shared__ __attribute__((aligned(32))) __bf16 Hbuf[M * 128];
  int base = blockIdx.x * M;
  int wave = threadIdx.x >> 5, lane = threadIdx.x & 31;
  for (int row = wave; row < M; row += 8) {
    int node = base + row;
    for (int k = lane; k < 128; k += 32) {
      float v = (node < nNodes) ? h[node * 128 + k] : 0.0f;
      Abuf[row * 128 + permk(k)] = (__bf16)v;
    }
  }
  __syncthreads();
  int kh = lane >> 4, mrow = lane & 15;

  // GEMM1: 128 -> 128, 8 waves x 1 N-tile, 4 M-subtiles each
  v8f acc[4];
#pragma unroll
  for (int mt = 0; mt < 4; ++mt) acc[mt] = (v8f){0, 0, 0, 0, 0, 0, 0, 0};
#pragma unroll
  for (int kt = 0; kt < 4; ++kt) {
    v16bf b = w1f[(kt * 8 + wave) * 32 + lane];
#pragma unroll
    for (int mt = 0; mt < 4; ++mt) {
      v16bf a = *(const v16bf*)&Abuf[(mt * 16 + mrow) * 128 + kt * 32 + kh * 16];
      acc[mt] = wmma_bf16(a, b, acc[mt]);
    }
  }
  float bias1 = b1[wave * 16 + mrow];
#pragma unroll
  for (int mt = 0; mt < 4; ++mt) {
#pragma unroll
    for (int r = 0; r < 8; ++r) {
      int m = mt * 16 + r + kh * 8;
      float v = acc[mt][r] + bias1;
      v = v > 0.0f ? v : 0.0f;
      Hbuf[m * 128 + permk(wave * 16 + mrow)] = (__bf16)v;
    }
  }
  __syncthreads();

  // GEMM2: 128 -> 64 (4 N-tiles); waves 0-3 take M-subtiles 0-1, waves 4-7 take 2-3
  int nt2 = wave & 3;
  int mtb = (wave >> 2) * 2;
  v8f acc2[2];
  acc2[0] = (v8f){0, 0, 0, 0, 0, 0, 0, 0};
  acc2[1] = (v8f){0, 0, 0, 0, 0, 0, 0, 0};
#pragma unroll
  for (int kt = 0; kt < 4; ++kt) {
    v16bf b = w2f[(kt * 4 + nt2) * 32 + lane];
#pragma unroll
    for (int j = 0; j < 2; ++j) {
      int mt = mtb + j;
      v16bf a = *(const v16bf*)&Hbuf[(mt * 16 + mrow) * 128 + kt * 32 + kh * 16];
      acc2[j] = wmma_bf16(a, b, acc2[j]);
    }
  }
  float bias2 = b2[nt2 * 16 + mrow];
#pragma unroll
  for (int j = 0; j < 2; ++j) {
    int mt = mtb + j;
#pragma unroll
    for (int r = 0; r < 8; ++r) {
      int m = mt * 16 + r + kh * 8;
      int node = base + m;
      if (node < nNodes) {
        float v = acc2[j][r] + bias2;
        out[node * 64 + nt2 * 16 + mrow] = v > 0.0f ? v : 0.0f;
      }
    }
  }
}

// ---------------------------------------------------------------------------
// SAGE neighbor aggregation (scatter sums)
// ---------------------------------------------------------------------------
__global__ void sage_agg_kernel(const float* __restrict__ h2, const float* __restrict__ mask,
                                const int* __restrict__ send, const int* __restrict__ recv,
                                float* __restrict__ nbr, float* __restrict__ cnt, int E) {
  int idx = blockIdx.x * blockDim.x + threadIdx.x;
  int e = idx >> 6, c = idx & 63;
  if (e >= E) return;
  float mk = mask[e];
  if (mk != 0.0f) {
    atomicAdd(&nbr[recv[e] * 64 + c], h2[send[e] * 64 + c] * mk);
    if (c == 0) atomicAdd(&cnt[recv[e]], mk);
  }
}

// ---------------------------------------------------------------------------
// SAGE dense (128->128) + L2 norm + relu + pooling atomics, 64 nodes/block
// ---------------------------------------------------------------------------
__global__ void sage_kernel(const float* __restrict__ h2, const float* __restrict__ nbr,
                            const float* __restrict__ cnt,
                            const v16bf* __restrict__ wf, const float* __restrict__ bias,
                            const int* __restrict__ seg,
                            float* __restrict__ pmax, float* __restrict__ psum,
                            float* __restrict__ npn, int nNodes) {
  constexpr int M = 64;
  __shared__ __attribute__((aligned(32))) __bf16 Abuf[M * 128];
  __shared__ float Gbuf[M * 128];
  __shared__ float rinv[M];
  int base = blockIdx.x * M;
  int wave = threadIdx.x >> 5, lane = threadIdx.x & 31;
  for (int row = wave; row < M; row += 8) {
    int node = base + row;
    for (int k = lane; k < 128; k += 32) {
      float v = 0.0f;
      if (node < nNodes) {
        if (k < 64) v = h2[node * 64 + k];
        else {
          float c = cnt[node]; c = c > 1.0f ? c : 1.0f;
          v = nbr[node * 64 + (k - 64)] / c;
        }
      }
      Abuf[row * 128 + permk(k)] = (__bf16)v;
    }
  }
  __syncthreads();
  int kh = lane >> 4, mrow = lane & 15;
  v8f acc[4];
#pragma unroll
  for (int mt = 0; mt < 4; ++mt) acc[mt] = (v8f){0, 0, 0, 0, 0, 0, 0, 0};
#pragma unroll
  for (int kt = 0; kt < 4; ++kt) {
    v16bf b = wf[(kt * 8 + wave) * 32 + lane];
#pragma unroll
    for (int mt = 0; mt < 4; ++mt) {
      v16bf a = *(const v16bf*)&Abuf[(mt * 16 + mrow) * 128 + kt * 32 + kh * 16];
      acc[mt] = wmma_bf16(a, b, acc[mt]);
    }
  }
  float bb = bias[wave * 16 + mrow];
#pragma unroll
  for (int mt = 0; mt < 4; ++mt) {
#pragma unroll
    for (int r = 0; r < 8; ++r) {
      int m = mt * 16 + r + kh * 8;
      Gbuf[m * 128 + wave * 16 + mrow] = acc[mt][r] + bb;
    }
  }
  __syncthreads();
  if (threadIdx.x < M) {
    float ss = 0.0f;
    for (int c = 0; c < 128; ++c) { float g = Gbuf[threadIdx.x * 128 + c]; ss += g * g; }
    ss = ss > 1e-12f ? ss : 1e-12f;
    rinv[threadIdx.x] = rsqrtf(ss);
  }
  __syncthreads();
  for (int idx = threadIdx.x; idx < M * 128; idx += 256) {
    int row = idx >> 7, c = idx & 127;
    int node = base + row;
    if (node < nNodes) {
      float v = Gbuf[idx] * rinv[row];
      v = v > 0.0f ? v : 0.0f;
      int s = seg[node];
      atomicMax((unsigned int*)&pmax[s * 128 + c], __float_as_uint(v));
      atomicAdd(&psum[s * 128 + c], v);
      if (c == 0) atomicAdd(&npn[s], 1.0f);
    }
  }
}

// ---------------------------------------------------------------------------
// Decoder (tiny: B=128 rows) — plain fp32 kernels
// ---------------------------------------------------------------------------
__global__ void build_z_kernel(const float* __restrict__ pmax, const float* __restrict__ psum,
                               const float* __restrict__ npn, float* __restrict__ z, int Bn) {
  int idx = blockIdx.x * blockDim.x + threadIdx.x;
  if (idx >= Bn * 384) return;
  int b = idx / 384, c = idx % 384;
  float v;
  if (c < 128) v = pmax[b * 128 + c];
  else if (c < 256) { float n = npn[b]; n = n > 1.0f ? n : 1.0f; v = psum[b * 128 + (c - 128)] / n; }
  else v = psum[b * 128 + (c - 256)];
  z[idx] = v;
}

// act: 0 = none, 1 = LeakyReLU(0.15)
__global__ void dense_kernel(const float* __restrict__ in, const float* __restrict__ W,
                             const float* __restrict__ bias, float* __restrict__ out,
                             int Bn, int K, int Nout, int act) {
  int idx = blockIdx.x * blockDim.x + threadIdx.x;
  if (idx >= Bn * Nout) return;
  int b = idx / Nout, o = idx % Nout;
  float s = bias[o];
  for (int k = 0; k < K; ++k) s += in[b * K + k] * W[k * Nout + o];
  if (act == 1) s = s > 0.0f ? s : 0.15f * s;
  out[idx] = s;
}

__global__ void bn_kernel(float* __restrict__ z, const float* __restrict__ g,
                          const float* __restrict__ be, const float* __restrict__ mu,
                          const float* __restrict__ var, int Bn, int C) {
  int idx = blockIdx.x * blockDim.x + threadIdx.x;
  if (idx >= Bn * C) return;
  int c = idx % C;
  z[idx] = (z[idx] - mu[c]) * rsqrtf(var[c] + 1e-3f) * g[c] + be[c];
}

__global__ void final_kernel(const float* __restrict__ loge, const float* __restrict__ ang,
                             const float* __restrict__ sig, const float* __restrict__ ascw,
                             const float* __restrict__ ascb, float* __restrict__ out, int Bn) {
  int b = blockIdx.x * blockDim.x + threadIdx.x;
  if (b >= Bn) return;
  float a0 = ang[b * 2 + 0], a1 = ang[b * 2 + 1];
  float t0 = a0 * ascw[0] + a1 * ascw[2] + ascb[0];
  float t1 = a0 * ascw[1] + a1 * ascw[3] + ascb[1];
  float z0 = 1.0f / (1.0f + expf(-t0));
  float z1 = 1.0f / (1.0f + expf(-t1));
  out[b * 5 + 0] = loge[b];
  out[b * 5 + 1] = z0 * 3.14159265358979f;
  out[b * 5 + 2] = z1 * 6.28318530717959f;
  out[b * 5 + 3] = fabsf(sig[b * 2 + 0]) + 1e-5f;
  out[b * 5 + 4] = fabsf(sig[b * 2 + 1]) + 1e-5f;
}

// ---------------------------------------------------------------------------
// Host: orchestration
// ---------------------------------------------------------------------------
static inline void* wsal(char** cur, size_t bytes) {
  void* p = (void*)*cur;
  *cur += (bytes + 255) & ~(size_t)255;
  return p;
}

extern "C" void kernel_launch(void* const* d_in, const int* in_sizes, int n_in,
                              void* d_out, int out_size, void* d_ws, size_t ws_size,
                              hipStream_t stream) {
  (void)n_in; (void)ws_size;
  const float* x       = (const float*)d_in[0];
  const int*   send    = (const int*)d_in[1];
  const int*   recv    = (const int*)d_in[2];
  const int*   seg     = (const int*)d_in[3];
  const float* msg0_w1 = (const float*)d_in[4];  const float* msg0_b1 = (const float*)d_in[5];
  const float* msg0_w2 = (const float*)d_in[6];  const float* msg0_b2 = (const float*)d_in[7];
  const float* msg1_w1 = (const float*)d_in[8];  const float* msg1_b1 = (const float*)d_in[9];
  const float* msg1_w2 = (const float*)d_in[10]; const float* msg1_b2 = (const float*)d_in[11];
  const float* upd_w1  = (const float*)d_in[12]; const float* upd_b1  = (const float*)d_in[13];
  const float* upd_w2  = (const float*)d_in[14]; const float* upd_b2  = (const float*)d_in[15];
  const float* sage_w  = (const float*)d_in[16]; const float* sage_b  = (const float*)d_in[17];
  const float* dec0_w  = (const float*)d_in[18]; const float* dec0_b  = (const float*)d_in[19];
  const float* bn0_g = (const float*)d_in[20]; const float* bn0_b = (const float*)d_in[21];
  const float* bn0_m = (const float*)d_in[22]; const float* bn0_v = (const float*)d_in[23];
  const float* dec1_w  = (const float*)d_in[24]; const float* dec1_b  = (const float*)d_in[25];
  const float* bn1_g = (const float*)d_in[26]; const float* bn1_b = (const float*)d_in[27];
  const float* bn1_m = (const float*)d_in[28]; const float* bn1_v = (const float*)d_in[29];
  const float* loge_w0 = (const float*)d_in[30]; const float* loge_b0 = (const float*)d_in[31];
  const float* loge_w1 = (const float*)d_in[32]; const float* loge_b1 = (const float*)d_in[33];
  const float* loge_wo = (const float*)d_in[34]; const float* loge_bo = (const float*)d_in[35];
  const float* ang_w0  = (const float*)d_in[36]; const float* ang_b0  = (const float*)d_in[37];
  const float* ang_w1  = (const float*)d_in[38]; const float* ang_b1  = (const float*)d_in[39];
  const float* ang_wo  = (const float*)d_in[40]; const float* ang_bo  = (const float*)d_in[41];
  const float* sig_w0  = (const float*)d_in[42]; const float* sig_b0  = (const float*)d_in[43];
  const float* sig_w1  = (const float*)d_in[44]; const float* sig_b1  = (const float*)d_in[45];
  const float* sig_wo  = (const float*)d_in[46]; const float* sig_bo  = (const float*)d_in[47];
  const float* asc_w   = (const float*)d_in[48]; const float* asc_b   = (const float*)d_in[49];

  const int N = in_sizes[0] / 6;
  const int E = in_sizes[1];
  const int Bn = out_size / 5;

  char* cur = (char*)d_ws;
  // Accumulator region (zeroed each call) — keep contiguous
  float* hA   = (float*)wsal(&cur, (size_t)N * 128 * 4);
  float* hB   = (float*)wsal(&cur, (size_t)N * 128 * 4);
  float* nbr  = (float*)wsal(&cur, (size_t)N * 64 * 4);
  float* cnt  = (float*)wsal(&cur, (size_t)N * 4);
  float* pmax = (float*)wsal(&cur, (size_t)Bn * 128 * 4);
  float* psum = (float*)wsal(&cur, (size_t)Bn * 128 * 4);
  float* npn  = (float*)wsal(&cur, (size_t)Bn * 4);
  size_t zbytes = (size_t)(cur - (char*)d_ws);
  // Non-accumulator scratch
  float* h2   = (float*)wsal(&cur, (size_t)N * 64 * 4);
  float* ef   = (float*)wsal(&cur, (size_t)E * 8 * 4);
  float* mask = (float*)wsal(&cur, (size_t)E * 4);
  float* z0   = (float*)wsal(&cur, (size_t)Bn * 384 * 4);
  float* z1   = (float*)wsal(&cur, (size_t)Bn * 1024 * 4);
  float* z2   = (float*)wsal(&cur, (size_t)Bn * 512 * 4);
  float* t64a = (float*)wsal(&cur, (size_t)Bn * 64 * 4);
  float* t64b = (float*)wsal(&cur, (size_t)Bn * 64 * 4);
  float* hloge = (float*)wsal(&cur, (size_t)Bn * 4);
  float* hang  = (float*)wsal(&cur, (size_t)Bn * 2 * 4);
  float* hsig  = (float*)wsal(&cur, (size_t)Bn * 2 * 4);
  // Packed bf16 weight fragments
  __bf16* f_m0w1 = (__bf16*)wsal(&cur, (size_t)1 * 16 * 512 * 2);   // Kpad32 ->256
  __bf16* f_m0w2 = (__bf16*)wsal(&cur, (size_t)8 * 8  * 512 * 2);   // 256->128
  __bf16* f_m1w1 = (__bf16*)wsal(&cur, (size_t)9 * 16 * 512 * 2);   // Kpad288->256
  __bf16* f_m1w2 = (__bf16*)wsal(&cur, (size_t)8 * 8  * 512 * 2);   // 256->128
  __bf16* f_uw1  = (__bf16*)wsal(&cur, (size_t)4 * 8  * 512 * 2);   // 128->128
  __bf16* f_uw2  = (__bf16*)wsal(&cur, (size_t)4 * 4  * 512 * 2);   // 128->64
  __bf16* f_sw   = (__bf16*)wsal(&cur, (size_t)4 * 8  * 512 * 2);   // 128->128

  const int TB = 256;
  auto g1 = [&](int n) { return dim3((n + TB - 1) / TB); };

  // 0) zero accumulators
  int zcnt = (int)(zbytes / 4);
  zero_kernel<<<g1(zcnt), TB, 0, stream>>>(hA, zcnt);

  // 1) pack weights to bf16 fragments
  pack_w_kernel<<<g1(1 * 16 * 512), TB, 0, stream>>>(msg0_w1, f_m0w1, 19, 256, 1, 16);
  pack_w_kernel<<<g1(8 * 8 * 512),  TB, 0, stream>>>(msg0_w2, f_m0w2, 256, 128, 8, 8);
  pack_w_kernel<<<g1(9 * 16 * 512), TB, 0, stream>>>(msg1_w1, f_m1w1, 263, 256, 9, 16);
  pack_w_kernel<<<g1(8 * 8 * 512),  TB, 0, stream>>>(msg1_w2, f_m1w2, 256, 128, 8, 8);
  pack_w_kernel<<<g1(4 * 8 * 512),  TB, 0, stream>>>(upd_w1,  f_uw1,  128, 128, 4, 8);
  pack_w_kernel<<<g1(4 * 4 * 512),  TB, 0, stream>>>(upd_w2,  f_uw2,  128, 64,  4, 4);
  pack_w_kernel<<<g1(4 * 8 * 512),  TB, 0, stream>>>(sage_w,  f_sw,   128, 128, 4, 8);

  // 2) edge features + mask
  efeat_kernel<<<g1(E), TB, 0, stream>>>(x, send, recv, ef, mask, E);

  // 3) two message-passing hops (WMMA)
  msg_hop_kernel<6, 32, 4><<<dim3((E + 63) / 64), TB, 0, stream>>>(
      x, ef, mask, send, recv, (const v16bf*)f_m0w1, msg0_b1,
      (const v16bf*)f_m0w2, msg0_b2, hA, E);
  msg_hop_kernel<128, 288, 2><<<dim3((E + 31) / 32), TB, 0, stream>>>(
      hA, ef, mask, send, recv, (const v16bf*)f_m1w1, msg1_b1,
      (const v16bf*)f_m1w2, msg1_b2, hB, E);

  // 4) node update MLP (WMMA)
  int ntiles64 = (N + 63) / 64;
  upd_kernel<<<dim3(ntiles64), TB, 0, stream>>>(hB, (const v16bf*)f_uw1, upd_b1,
                                                (const v16bf*)f_uw2, upd_b2, h2, N);

  // 5) SAGE aggregate + dense + norm + pooling (WMMA)
  sage_agg_kernel<<<g1(E * 64), TB, 0, stream>>>(h2, mask, send, recv, nbr, cnt, E);
  sage_kernel<<<dim3(ntiles64), TB, 0, stream>>>(h2, nbr, cnt, (const v16bf*)f_sw, sage_b,
                                                 seg, pmax, psum, npn, N);

  // 6) decoder
  build_z_kernel<<<g1(Bn * 384), TB, 0, stream>>>(pmax, psum, npn, z0, Bn);
  dense_kernel<<<g1(Bn * 1024), TB, 0, stream>>>(z0, dec0_w, dec0_b, z1, Bn, 384, 1024, 1);
  bn_kernel<<<g1(Bn * 1024), TB, 0, stream>>>(z1, bn0_g, bn0_b, bn0_m, bn0_v, Bn, 1024);
  dense_kernel<<<g1(Bn * 512), TB, 0, stream>>>(z1, dec1_w, dec1_b, z2, Bn, 1024, 512, 1);
  bn_kernel<<<g1(Bn * 512), TB, 0, stream>>>(z2, bn1_g, bn1_b, bn1_m, bn1_v, Bn, 512);

  // 7) heads
  dense_kernel<<<g1(Bn * 64), TB, 0, stream>>>(z2, loge_w0, loge_b0, t64a, Bn, 512, 64, 0);
  dense_kernel<<<g1(Bn * 64), TB, 0, stream>>>(t64a, loge_w1, loge_b1, t64b, Bn, 64, 64, 0);
  dense_kernel<<<g1(Bn * 1), TB, 0, stream>>>(t64b, loge_wo, loge_bo, hloge, Bn, 64, 1, 0);
  dense_kernel<<<g1(Bn * 64), TB, 0, stream>>>(z2, ang_w0, ang_b0, t64a, Bn, 512, 64, 0);
  dense_kernel<<<g1(Bn * 64), TB, 0, stream>>>(t64a, ang_w1, ang_b1, t64b, Bn, 64, 64, 0);
  dense_kernel<<<g1(Bn * 2), TB, 0, stream>>>(t64b, ang_wo, ang_bo, hang, Bn, 64, 2, 0);
  dense_kernel<<<g1(Bn * 64), TB, 0, stream>>>(z2, sig_w0, sig_b0, t64a, Bn, 512, 64, 0);
  dense_kernel<<<g1(Bn * 64), TB, 0, stream>>>(t64a, sig_w1, sig_b1, t64b, Bn, 64, 64, 0);
  dense_kernel<<<g1(Bn * 2), TB, 0, stream>>>(t64b, sig_wo, sig_bo, hsig, Bn, 64, 2, 0);

  // 8) final combine
  final_kernel<<<g1(Bn), TB, 0, stream>>>(hloge, hang, hsig, asc_w, asc_b, (float*)d_out, Bn);
}
